// RPN_62749472194571
// MI455X (gfx1250) — matrix operand order repaired and checked by reading, not compiled, not run on hardware
//
#include <hip/hip_runtime.h>
#include <cstddef>

// ---------------------------------------------------------------------------
// Types for WMMA
// ---------------------------------------------------------------------------
typedef __bf16 v16bf __attribute__((ext_vector_type(16)));
typedef float  v8f   __attribute__((ext_vector_type(8)));
typedef unsigned int u32x4 __attribute__((ext_vector_type(4)));

union Frag32B {
    v16bf v;
    u32x4 q[2];
    unsigned int u[8];
};

#define CIN   256
#define COUT  256
#define NLVL  5
#define TOTC  4768          // 1000*4 + 768 candidates per image
#define NW    149           // ceil(4768/32) == 149 exactly
#define PADN  8192          // bitonic pad
#define NEGS  (-1.0e4f)
#define IMGSZ 1024.0f
#define SCALE_CLAMP 4.135166556742356f   // log(1000/16)

__device__ __forceinline__ float bf2f(unsigned short h) {
    return __uint_as_float(((unsigned int)h) << 16);
}
__device__ __forceinline__ unsigned short f2bf(float f) {
    unsigned int u = __float_as_uint(f);
    u = u + 0x7FFFu + ((u >> 16) & 1u);   // round-to-nearest-even
    return (unsigned short)(u >> 16);
}
__device__ __forceinline__ unsigned int fkey(float f) {
    unsigned int u = __float_as_uint(f);
    return (u & 0x80000000u) ? ~u : (u | 0x80000000u);
}

// ---------------------------------------------------------------------------
// Kernel 1: NCHW f32 -> NHWC bf16 (LDS tiled transpose), per level, 2 images
// ---------------------------------------------------------------------------
__global__ __launch_bounds__(256)
void k_nchw_to_nhwc(const float* __restrict__ in, unsigned short* __restrict__ out, int HW)
{
    __shared__ float tile[32][33];
    const int tx = threadIdx.x;        // 0..31
    const int ty = threadIdx.y;        // 0..7
    const int c0 = blockIdx.y * 32;    // channel tile (grid.y = 8)
    const int p0 = blockIdx.x * 32;    // pixel tile
    const int img = blockIdx.z;
    const float* src = in + (size_t)img * CIN * HW;
    unsigned short* dst = out + (size_t)img * HW * CIN;
#pragma unroll
    for (int i = ty; i < 32; i += 8) {
        int p = p0 + tx;
        tile[i][tx] = (p < HW) ? src[(size_t)(c0 + i) * HW + p] : 0.0f;
    }
    __syncthreads();
#pragma unroll
    for (int i = ty; i < 32; i += 8) {
        int p = p0 + i;
        if (p < HW) dst[(size_t)p * CIN + (c0 + tx)] = f2bf(tile[tx][i]);
    }
}

// ---------------------------------------------------------------------------
// Kernel 2: repack w_conv [O=256][I=256][3][3] f32 into per-lane B fragments:
//   frag index = (kIdx*16 + nTile)*32 + lane, each frag lane holds 16 bf16
//   kIdx = tap*8 + kc (K chunk of 32), element j -> K_local = (lane>>4)*16 + j
// ---------------------------------------------------------------------------
__global__ __launch_bounds__(256)
void k_pack_weights(const float* __restrict__ wconv, unsigned short* __restrict__ wpack)
{
    int gid = blockIdx.x * 256 + threadIdx.x;   // over 72*16*32 = 36864
    if (gid >= 72 * 16 * 32) return;
    int lane = gid & 31;
    int nt   = (gid >> 5) & 15;
    int kIdx = gid >> 9;            // 0..71
    int tap  = kIdx >> 3;
    int kc   = kIdx & 7;
    int n    = nt * 16 + (lane & 15);
    int kh   = (lane >> 4) * 16;
    unsigned short* dst = wpack + (size_t)gid * 16;
#pragma unroll
    for (int j = 0; j < 16; ++j) {
        int ci = kc * 32 + kh + j;
        float v = wconv[(((size_t)n * CIN + ci) * 3 + tap / 3) * 3 + (tap % 3)];
        dst[j] = f2bf(v);
    }
}

// ---------------------------------------------------------------------------
// Kernel 2b: pack head weights [15 ch + 1 zero pad][256] -> B fragments
//   kIdx = kc (K chunk of 32, 8 total); one n-tile of 16 channels
// ---------------------------------------------------------------------------
__global__ __launch_bounds__(256)
void k_pack_head(const float* __restrict__ w_obj, const float* __restrict__ w_delta,
                 unsigned short* __restrict__ wpackH)
{
    int gid = threadIdx.x;          // 256 = 8 kc * 32 lanes
    int lane = gid & 31;
    int kc   = gid >> 5;
    int c    = lane & 15;           // output channel (15 = pad)
    int kh   = (lane >> 4) * 16;
    unsigned short* dst = wpackH + (size_t)gid * 16;
#pragma unroll
    for (int j = 0; j < 16; ++j) {
        int ci = kc * 32 + kh + j;
        float v = (c < 3)  ? w_obj[c * 256 + ci]
                : (c < 15) ? w_delta[(c - 3) * 256 + ci]
                : 0.0f;
        dst[j] = f2bf(v);
    }
}

// ---------------------------------------------------------------------------
// Kernel 3: conv3x3 + bias + ReLU via implicit GEMM WMMA bf16
//   block = 256 thr (8 waves); block tile 64 pixels x 256 couts
//   wave: mGroup (0..1) x nGroup (0..3): 2 M-tiles x 4 N-tiles, 8 accums
// ---------------------------------------------------------------------------
__global__ __launch_bounds__(256)
void k_conv3x3_wmma(const unsigned short* __restrict__ act,   // [P][256] NHWC bf16
                    const unsigned short* __restrict__ wpack, // packed B frags
                    const float* __restrict__ bias,
                    unsigned short* __restrict__ tout,        // [P][256] bf16
                    int H, int W, int P)
{
    const int lane  = threadIdx.x & 31;
    const int wave  = threadIdx.x >> 5;
    const int mGrp  = wave & 1;
    const int nGrp  = wave >> 1;
    const int pixB  = blockIdx.x * 64;
    const int half  = lane >> 4;
    const int mrow  = lane & 15;
    const int HWl   = H * W;
    const int ntB   = nGrp * 4;   // first global n-tile for this wave

    int  rImg[2], rY[2], rX[2];
#pragma unroll
    for (int mt = 0; mt < 2; ++mt) {
        int p = pixB + (mGrp * 2 + mt) * 16 + mrow;   // P is a multiple of 64
        rImg[mt] = p / HWl;
        int rem  = p - rImg[mt] * HWl;
        rY[mt] = rem / W;
        rX[mt] = rem - rY[mt] * W;
    }

    v8f acc[2][4];
#pragma unroll
    for (int mt = 0; mt < 2; ++mt)
#pragma unroll
        for (int nt = 0; nt < 4; ++nt)
#pragma unroll
            for (int e = 0; e < 8; ++e) acc[mt][nt][e] = 0.0f;

    for (int tap = 0; tap < 9; ++tap) {
        const int dy = tap / 3 - 1, dx = tap % 3 - 1;
        const unsigned short* aRow[2];
        bool av[2];
#pragma unroll
        for (int mt = 0; mt < 2; ++mt) {
            int ny = rY[mt] + dy, nx = rX[mt] + dx;
            av[mt] = ((unsigned)ny < (unsigned)H) && ((unsigned)nx < (unsigned)W);
            aRow[mt] = act + ((size_t)((rImg[mt] * H + ny) * W + nx)) * CIN + half * 16;
        }
        for (int kc = 0; kc < 8; ++kc) {
            Frag32B a[2];
#pragma unroll
            for (int mt = 0; mt < 2; ++mt) {
                if (av[mt]) {
                    const unsigned short* p = aRow[mt] + kc * 32;
                    a[mt].q[0] = *(const u32x4*)(p);
                    a[mt].q[1] = *(const u32x4*)(p + 8);
                } else {
#pragma unroll
                    for (int j = 0; j < 8; ++j) a[mt].u[j] = 0u;
                }
            }
            const int kIdx = tap * 8 + kc;
            Frag32B b[4];
#pragma unroll
            for (int nt = 0; nt < 4; ++nt) {
                const unsigned short* p =
                    wpack + ((size_t)((kIdx * 16 + ntB + nt) * 32 + lane)) * 16;
                b[nt].q[0] = *(const u32x4*)(p);
                b[nt].q[1] = *(const u32x4*)(p + 8);
            }
#pragma unroll
            for (int mt = 0; mt < 2; ++mt)
#pragma unroll
                for (int nt = 0; nt < 4; ++nt)
                    acc[mt][nt] = __builtin_amdgcn_wmma_f32_16x16x32_bf16(
                        false, a[mt].v, false, b[nt].v,
                        (short)0, acc[mt][nt], false, false);
        }
    }

    // epilogue: bias + ReLU + bf16 store (C layout: VGPR r -> M=r+8*half, N=lane&15)
#pragma unroll
    for (int mt = 0; mt < 2; ++mt) {
#pragma unroll
        for (int nt = 0; nt < 4; ++nt) {
            const int coutBase = (ntB + nt) * 16;
            const float bia = bias[coutBase + mrow];
#pragma unroll
            for (int r = 0; r < 8; ++r) {
                int p = pixB + (mGrp * 2 + mt) * 16 + r + half * 8;
                float vv = acc[mt][nt][r] + bia;
                vv = vv > 0.0f ? vv : 0.0f;
                tout[(size_t)p * COUT + coutBase + mrow] = f2bf(vv);
            }
        }
    }
}

// ---------------------------------------------------------------------------
// Kernel 4: 1x1 heads via WMMA: [P x 256] x [256 x 16(=15+pad)]
//   block = 256 thr (8 waves); wave -> 16 pixels, K loop 8 x 32
// ---------------------------------------------------------------------------
__global__ __launch_bounds__(256)
void k_head_wmma(const unsigned short* __restrict__ t,
                 const unsigned short* __restrict__ wpackH,
                 const float* __restrict__ b_obj, const float* __restrict__ b_delta,
                 float* __restrict__ scores, float* __restrict__ deltas, int P)
{
    const int lane = threadIdx.x & 31;
    const int wave = threadIdx.x >> 5;
    const int half = lane >> 4;
    const int mrow = lane & 15;
    const int pixB = blockIdx.x * 128 + wave * 16;   // P multiple of 128

    v8f acc;
#pragma unroll
    for (int e = 0; e < 8; ++e) acc[e] = 0.0f;

    const unsigned short* aBase = t + (size_t)(pixB + mrow) * 256 + half * 16;
#pragma unroll
    for (int kc = 0; kc < 8; ++kc) {
        Frag32B a, b;
        const unsigned short* ap = aBase + kc * 32;
        a.q[0] = *(const u32x4*)(ap);
        a.q[1] = *(const u32x4*)(ap + 8);
        const unsigned short* bp = wpackH + ((size_t)(kc * 32 + lane)) * 16;
        b.q[0] = *(const u32x4*)(bp);
        b.q[1] = *(const u32x4*)(bp + 8);
        acc = __builtin_amdgcn_wmma_f32_16x16x32_bf16(
            false, a.v, false, b.v, (short)0, acc, false, false);
    }

    const int c = mrow;   // output channel of this lane's column
    float bia = (c < 3) ? b_obj[c] : (c < 15 ? b_delta[c - 3] : 0.0f);
#pragma unroll
    for (int r = 0; r < 8; ++r) {
        int p = pixB + r + half * 8;
        float v = acc[r] + bia;
        if (c < 3) {
            scores[(size_t)p * 3 + c] = v;
        } else if (c < 15) {
            int ch = c - 3, aa = ch >> 2, cmp = ch & 3;
            deltas[((size_t)p * 3 + aa) * 4 + cmp] = v;
        }
    }
}

// ---------------------------------------------------------------------------
// Kernel 5: per-(level,image) top-k via 4-pass radix select (block per image)
// ---------------------------------------------------------------------------
__global__ __launch_bounds__(1024)
void k_topk(const float* __restrict__ scores, int M, int k,
            int* __restrict__ selIdx, float* __restrict__ selScore)
{
    const float* sc = scores + (size_t)blockIdx.x * M;
    int*   oIdx = selIdx   + blockIdx.x * 1000;
    float* oSc  = selScore + blockIdx.x * 1000;
    __shared__ unsigned int hist[256];
    __shared__ unsigned int sPrefix, sK, cGt, cEq;
    if (threadIdx.x == 0) { sPrefix = 0u; sK = (unsigned)k; }
    for (int pass = 3; pass >= 0; --pass) {
        for (int i = threadIdx.x; i < 256; i += 1024) hist[i] = 0u;
        __syncthreads();
        const unsigned int prefix = sPrefix;
        const unsigned int shift  = (unsigned)pass * 8u;
        const unsigned int pmask  = (pass == 3) ? 0u : (0xFFFFFFFFu << ((pass + 1) * 8));
        for (int i = threadIdx.x; i < M; i += 1024) {
            unsigned int key = fkey(sc[i]);
            if ((key & pmask) == (prefix & pmask))
                atomicAdd(&hist[(key >> shift) & 255u], 1u);
        }
        __syncthreads();
        if (threadIdx.x == 0) {
            unsigned int rem = sK;
            unsigned int b = 255u;
            for (;;) {
                unsigned int c = hist[b];
                if (c >= rem || b == 0u) break;
                rem -= c; --b;
            }
            sPrefix = prefix | (b << shift);
            sK = rem;
        }
        __syncthreads();
    }
    if (threadIdx.x == 0) { cGt = 0u; cEq = 0u; }
    __syncthreads();
    const unsigned int T = sPrefix;
    const unsigned int quotaEq = sK;
    const unsigned int nGt = (unsigned)k - quotaEq;
    for (int i = threadIdx.x; i < M; i += 1024) {
        unsigned int key = fkey(sc[i]);
        if (key > T) {
            unsigned int pos = atomicAdd(&cGt, 1u);
            oIdx[pos] = i; oSc[pos] = sc[i];
        } else if (key == T) {
            unsigned int pos = atomicAdd(&cEq, 1u);
            if (pos < quotaEq) { oIdx[nGt + pos] = i; oSc[nGt + pos] = sc[i]; }
        }
    }
}

// ---------------------------------------------------------------------------
// Kernel 6: anchors + apply_deltas + clip + valid + level-offset boxes
// ---------------------------------------------------------------------------
struct LvlMeta {
    int W[5]; int HWa[5]; int k[5]; int off[5];
    float stride[5]; float asize[5];
    long long dOff[5];   // float-element offset of each level's delta buffer
};

__global__ __launch_bounds__(256)
void k_build_boxes(const int* __restrict__ selIdx, const float* __restrict__ selScore,
                   const float* __restrict__ deltasBase, LvlMeta meta,
                   float* __restrict__ boxes, float* __restrict__ oboxes,
                   float* __restrict__ bscores, int* __restrict__ bvalid)
{
    int gid = blockIdx.x * 256 + threadIdx.x;
    if (gid >= 2 * TOTC) return;
    int img  = gid / TOTC;
    int slot = gid - img * TOTC;
    int l = 0;
    while (l < 4 && slot >= meta.off[l] + meta.k[l]) ++l;
    int j = slot - meta.off[l];
    int cand = selIdx[(l * 2 + img) * 1000 + j];
    float score = selScore[(l * 2 + img) * 1000 + j];
    int pix = cand / 3;
    int a   = cand - pix * 3;
    int W   = meta.W[l];
    int x   = pix % W;
    int y   = pix / W;
    const float ratios[3] = {0.5f, 1.0f, 2.0f};
    float r  = ratios[a];
    float aw = meta.asize[l] / sqrtf(r);
    float ah = aw * r;
    float cx = (float)x * meta.stride[l];
    float cy = (float)y * meta.stride[l];
    const float* d = deltasBase + meta.dOff[l]
                   + ((size_t)(img * meta.HWa[l] + pix) * 3 + a) * 4;
    float dxv = d[0], dyv = d[1];
    float dwv = fminf(d[2], SCALE_CLAMP);
    float dhv = fminf(d[3], SCALE_CLAMP);
    float pcx = dxv * aw + cx;
    float pcy = dyv * ah + cy;
    float pw  = __expf(dwv) * aw;
    float ph  = __expf(dhv) * ah;
    float b0 = pcx - 0.5f * pw, b1 = pcy - 0.5f * ph;
    float b2 = pcx + 0.5f * pw, b3 = pcy + 0.5f * ph;
    b0 = fminf(fmaxf(b0, 0.0f), IMGSZ);
    b1 = fminf(fmaxf(b1, 0.0f), IMGSZ);
    b2 = fminf(fmaxf(b2, 0.0f), IMGSZ);
    b3 = fminf(fmaxf(b3, 0.0f), IMGSZ);
    int vld = ((b2 - b0) > 0.0f) && ((b3 - b1) > 0.0f);
    float offv = (float)l * (IMGSZ * 4.0f);
    size_t o = (size_t)gid * 4;
    boxes[o + 0] = b0; boxes[o + 1] = b1; boxes[o + 2] = b2; boxes[o + 3] = b3;
    oboxes[o + 0] = b0 + offv; oboxes[o + 1] = b1 + offv;
    oboxes[o + 2] = b2 + offv; oboxes[o + 3] = b3 + offv;
    bscores[gid] = score;
    bvalid[gid]  = vld;
}

// ---------------------------------------------------------------------------
// Kernel 7: per-image descending bitonic sort (8192 pad) in LDS, then gather
// ---------------------------------------------------------------------------
__global__ __launch_bounds__(1024)
void k_sort(const float* __restrict__ bscores, const float* __restrict__ boxes,
            const float* __restrict__ oboxes, const int* __restrict__ bvalid,
            float* __restrict__ sScore, float* __restrict__ sBoxes,
            float* __restrict__ sObox, int* __restrict__ sValid)
{
    __shared__ float key[PADN];
    __shared__ int   idx[PADN];
    const int img = blockIdx.x;
    for (int i = threadIdx.x; i < PADN; i += 1024) {
        key[i] = (i < TOTC) ? bscores[img * TOTC + i] : -3.0e38f;
        idx[i] = i;
    }
    __syncthreads();
    for (int kk = 2; kk <= PADN; kk <<= 1) {
        for (int jj = kk >> 1; jj > 0; jj >>= 1) {
            for (int i = threadIdx.x; i < PADN; i += 1024) {
                int ixj = i ^ jj;
                if (ixj > i) {
                    bool upper = (i & kk) != 0;
                    float ai = key[i], aj = key[ixj];
                    bool sw = upper ? (ai > aj) : (ai < aj);   // overall descending
                    if (sw) {
                        key[i] = aj; key[ixj] = ai;
                        int t = idx[i]; idx[i] = idx[ixj]; idx[ixj] = t;
                    }
                }
            }
            __syncthreads();
        }
    }
    for (int i = threadIdx.x; i < TOTC; i += 1024) {
        int s = idx[i];
        sScore[img * TOTC + i] = key[i];
        sValid[img * TOTC + i] = bvalid[img * TOTC + s];
#pragma unroll
        for (int c = 0; c < 4; ++c) {
            sBoxes[((size_t)img * TOTC + i) * 4 + c] = boxes[((size_t)img * TOTC + s) * 4 + c];
            sObox [((size_t)img * TOTC + i) * 4 + c] = oboxes[((size_t)img * TOTC + s) * 4 + c];
        }
    }
}

// ---------------------------------------------------------------------------
// Kernel 8: suppression bitmask (IoU > 0.7, j < i), one row per block
// ---------------------------------------------------------------------------
__global__ __launch_bounds__(160)
void k_iou_mask(const float* __restrict__ sObox, unsigned int* __restrict__ mask)
{
    const int row = blockIdx.x;
    const int img = blockIdx.y;
    const float* B = sObox + (size_t)img * TOTC * 4;
    const float x0 = B[row * 4 + 0], y0 = B[row * 4 + 1];
    const float x1 = B[row * 4 + 2], y1 = B[row * 4 + 3];
    const float areaI = (x1 - x0) * (y1 - y0);
    const int w = threadIdx.x;
    if (w >= NW) return;
    unsigned int bits = 0u;
    const int jBase = w * 32;
    for (int b = 0; b < 32; ++b) {
        int j = jBase + b;
        if (j >= row) break;
        float u0 = B[j * 4 + 0], v0 = B[j * 4 + 1];
        float u1 = B[j * 4 + 2], v1 = B[j * 4 + 3];
        float areaJ = (u1 - u0) * (v1 - v0);
        float lx = fmaxf(x0, u0), ly = fmaxf(y0, v0);
        float rx = fminf(x1, u1), ry = fminf(y1, v1);
        float iw = fmaxf(rx - lx, 0.0f), ih = fmaxf(ry - ly, 0.0f);
        float inter = iw * ih;
        float iou = inter / (areaI + areaJ - inter + 1e-9f);
        if (iou > 0.7f) bits |= (1u << b);
    }
    mask[((size_t)img * TOTC + row) * NW + w] = bits;
}

// ---------------------------------------------------------------------------
// Kernel 9: sequential greedy NMS scan + stable keep-first emission
// ---------------------------------------------------------------------------
__global__ __launch_bounds__(256)
void k_greedy(const unsigned int* __restrict__ mask, const int* __restrict__ sValid,
              const float* __restrict__ sScore, const float* __restrict__ sBoxes,
              float* __restrict__ out)
{
    const int img = blockIdx.x;
    __shared__ unsigned int keepW[NW];
    __shared__ int flag;
    for (int i = threadIdx.x; i < NW; i += 256) keepW[i] = 0u;
    __syncthreads();
    for (int i = 0; i < TOTC; ++i) {
        if (threadIdx.x == 0) flag = 0;
        __syncthreads();
        const unsigned int* mrow = mask + ((size_t)img * TOTC + i) * NW;
        const int nw = (i >> 5) + 1;
        for (int w = threadIdx.x; w < nw; w += 256)
            if (keepW[w] & mrow[w]) flag = 1;
        __syncthreads();
        if (threadIdx.x == 0) {
            if (sValid[img * TOTC + i] && !flag)
                keepW[i >> 5] |= (1u << (i & 31));
        }
        __syncthreads();
    }
    if (threadIdx.x == 0) {
        int pos = 0;
        for (int i = 0; i < TOTC && pos < 1000; ++i) {
            if (keepW[i >> 5] & (1u << (i & 31))) {
                float* o = out + ((size_t)img * 1000 + pos) * 5;
                for (int c = 0; c < 4; ++c) o[c] = sBoxes[((size_t)img * TOTC + i) * 4 + c];
                o[4] = sScore[img * TOTC + i];
                ++pos;
            }
        }
        for (int i = 0; i < TOTC && pos < 1000; ++i) {
            if (!(keepW[i >> 5] & (1u << (i & 31)))) {
                float* o = out + ((size_t)img * 1000 + pos) * 5;
                for (int c = 0; c < 4; ++c) o[c] = sBoxes[((size_t)img * TOTC + i) * 4 + c];
                o[4] = NEGS;
                ++pos;
            }
        }
    }
}

// ---------------------------------------------------------------------------
// Host launch
// ---------------------------------------------------------------------------
extern "C" void kernel_launch(void* const* d_in, const int* in_sizes, int n_in,
                              void* d_out, int out_size, void* d_ws, size_t ws_size,
                              hipStream_t stream)
{
    const float* feats[5];
    for (int i = 0; i < 5; ++i) feats[i] = (const float*)d_in[i];
    const float* w_conv  = (const float*)d_in[5];
    const float* b_conv  = (const float*)d_in[6];
    const float* w_obj   = (const float*)d_in[7];
    const float* b_obj   = (const float*)d_in[8];
    const float* w_delta = (const float*)d_in[9];
    const float* b_delta = (const float*)d_in[10];

    static const int   Hs[5]      = {256, 128, 64, 32, 16};
    static const float strides[5] = {4.f, 8.f, 16.f, 32.f, 64.f};
    static const float sizesA[5]  = {32.f, 64.f, 128.f, 256.f, 512.f};

    int HW[5], P[5], Ksel[5], SO[5];
    int tot = 0;
    for (int l = 0; l < 5; ++l) {
        HW[l] = Hs[l] * Hs[l];
        P[l]  = 2 * HW[l];
        Ksel[l] = (3 * HW[l] < 1000) ? 3 * HW[l] : 1000;
        SO[l] = tot;
        tot += Ksel[l];
    }

    // workspace carve-up
    size_t cur = 0;
    auto alloc = [&](size_t bytes) -> size_t {
        cur = (cur + 255) & ~(size_t)255;
        size_t o = cur; cur += bytes; return o;
    };
    size_t actOff[5], tOff[5], scOff[5], dlOff[5];
    size_t oWpack  = alloc((size_t)72 * 16 * 32 * 16 * 2);
    size_t oWpackH = alloc((size_t)8 * 32 * 16 * 2);
    for (int l = 0; l < 5; ++l) actOff[l] = alloc((size_t)P[l] * CIN * 2);
    for (int l = 0; l < 5; ++l) tOff[l]   = alloc((size_t)P[l] * COUT * 2);
    for (int l = 0; l < 5; ++l) scOff[l]  = alloc((size_t)P[l] * 3 * 4);
    size_t dlBase = 0;
    {
        size_t first = alloc((size_t)P[0] * 12 * 4);
        dlOff[0] = first; dlBase = first;
        for (int l = 1; l < 5; ++l) dlOff[l] = alloc((size_t)P[l] * 12 * 4);
    }
    size_t oSelIdx = alloc((size_t)5 * 2 * 1000 * 4);
    size_t oSelSc  = alloc((size_t)5 * 2 * 1000 * 4);
    size_t oBoxes  = alloc((size_t)2 * TOTC * 4 * 4);
    size_t oObox   = alloc((size_t)2 * TOTC * 4 * 4);
    size_t oBsc    = alloc((size_t)2 * TOTC * 4);
    size_t oBval   = alloc((size_t)2 * TOTC * 4);
    size_t oSsc    = alloc((size_t)2 * TOTC * 4);
    size_t oSbox   = alloc((size_t)2 * TOTC * 4 * 4);
    size_t oSobox  = alloc((size_t)2 * TOTC * 4 * 4);
    size_t oSval   = alloc((size_t)2 * TOTC * 4);
    size_t oMask   = alloc((size_t)2 * TOTC * NW * 4);

    char* ws = (char*)d_ws;
    unsigned short* wpack  = (unsigned short*)(ws + oWpack);
    unsigned short* wpackH = (unsigned short*)(ws + oWpackH);

    // 1) pack weights (conv + heads)
    k_pack_weights<<<(72 * 16 * 32 + 255) / 256, 256, 0, stream>>>(w_conv, wpack);
    k_pack_head<<<1, 256, 0, stream>>>(w_obj, w_delta, wpackH);

    // 2) NCHW -> NHWC bf16
    for (int l = 0; l < 5; ++l) {
        dim3 g((HW[l] + 31) / 32, 8, 2), b(32, 8, 1);
        k_nchw_to_nhwc<<<g, b, 0, stream>>>(feats[l], (unsigned short*)(ws + actOff[l]), HW[l]);
    }

    // 3) conv3x3 + ReLU (WMMA)
    for (int l = 0; l < 5; ++l) {
        k_conv3x3_wmma<<<(P[l] + 63) / 64, 256, 0, stream>>>(
            (const unsigned short*)(ws + actOff[l]), wpack, b_conv,
            (unsigned short*)(ws + tOff[l]), Hs[l], Hs[l], P[l]);
    }

    // 4) 1x1 heads (WMMA)
    for (int l = 0; l < 5; ++l) {
        k_head_wmma<<<P[l] / 128, 256, 0, stream>>>(
            (const unsigned short*)(ws + tOff[l]), wpackH, b_obj, b_delta,
            (float*)(ws + scOff[l]), (float*)(ws + dlOff[l]), P[l]);
    }

    // 5) per-level top-k
    for (int l = 0; l < 5; ++l) {
        k_topk<<<2, 1024, 0, stream>>>(
            (const float*)(ws + scOff[l]), 3 * HW[l], Ksel[l],
            (int*)(ws + oSelIdx) + l * 2000, (float*)(ws + oSelSc) + l * 2000);
    }

    // 6) decode boxes
    LvlMeta meta;
    for (int l = 0; l < 5; ++l) {
        meta.W[l] = Hs[l]; meta.HWa[l] = HW[l]; meta.k[l] = Ksel[l]; meta.off[l] = SO[l];
        meta.stride[l] = strides[l]; meta.asize[l] = sizesA[l];
        meta.dOff[l] = (long long)((dlOff[l] - dlBase) / 4);
    }
    k_build_boxes<<<(2 * TOTC + 255) / 256, 256, 0, stream>>>(
        (const int*)(ws + oSelIdx), (const float*)(ws + oSelSc),
        (const float*)(ws + dlBase), meta,
        (float*)(ws + oBoxes), (float*)(ws + oObox),
        (float*)(ws + oBsc), (int*)(ws + oBval));

    // 7) sort per image
    k_sort<<<2, 1024, 0, stream>>>(
        (const float*)(ws + oBsc), (const float*)(ws + oBoxes),
        (const float*)(ws + oObox), (const int*)(ws + oBval),
        (float*)(ws + oSsc), (float*)(ws + oSbox),
        (float*)(ws + oSobox), (int*)(ws + oSval));

    // 8) IoU suppression bitmask
    {
        dim3 g(TOTC, 2);
        k_iou_mask<<<g, 160, 0, stream>>>((const float*)(ws + oSobox),
                                          (unsigned int*)(ws + oMask));
    }

    // 9) greedy NMS + emit [2,1000,5]
    k_greedy<<<2, 256, 0, stream>>>(
        (const unsigned int*)(ws + oMask), (const int*)(ws + oSval),
        (const float*)(ws + oSsc), (const float*)(ws + oSbox),
        (float*)d_out);
}
